// AutoEncoder_21543555957319
// MI455X (gfx1250) — matrix operand order repaired and checked by reading, not compiled, not run on hardware
//
#include <hip/hip_runtime.h>

// ---------------------------------------------------------------------------
// Seq2seq LSTM autoencoder forward, CDNA5 (gfx1250) wave32 + WMMA bf16.
// B=64, T=256, V=8000, E=512, H=1024.
// x/h stored pre-packed in WMMA A-fragment bf16 layout, weights pre-packed in
// B-fragment layout.  Gates GEMM: 2 N-tiles per wave (A reuse, WMMA ILP).
// Logits GEMM: B stream double-buffered through LDS with
// global_load_async_to_lds_b128 + s_wait_asynccnt (CDNA5 async path).
// ---------------------------------------------------------------------------

#define BB   64
#define TT   256
#define VV   8000
#define EE   512
#define HH   1024
#define G4H  (4 * HH)   // 4096

typedef __attribute__((ext_vector_type(16))) __bf16 v16bf;
typedef __attribute__((ext_vector_type(8)))  float  v8f;

union Frag16 {
    unsigned int u[8];
    v16bf v;
};

// Round-to-nearest-even f32 -> bf16, packed pair into one dword (p=0 in [15:0]).
__device__ __forceinline__ unsigned int pk_bf16(float lo, float hi) {
    unsigned int a = __float_as_uint(lo);
    unsigned int b = __float_as_uint(hi);
    a = (a + 0x7FFFu + ((a >> 16) & 1u)) >> 16;
    b = (b + 0x7FFFu + ((b >> 16) & 1u)) >> 16;
    return (a & 0xFFFFu) | (b << 16);
}

// Packed-fragment dword address.  Memory layout (A and B buffers):
//   dword = tile*256 + (j>>2)*128 + lane*4 + (j&3)
// A fragment (16x32, M=row, K=col): lane = (M%16) + 16*h where
//   kk = K%32, h = (kk>>3)&1, j = ((kk&16)>>2) + ((kk&7)>>1), p = kk&1.
__device__ __forceinline__ long long a_frag_dword(int m, int k, int nKt) {
    int kt   = k >> 5;
    int kk   = k & 31;
    int hh   = (kk >> 3) & 1;
    int j    = ((kk & 16) >> 2) + ((kk & 7) >> 1);
    int lane = (m & 15) + (hh << 4);
    long long tile = (long long)(m >> 4) * nKt + kt;
    return tile * 256 + ((j >> 2) << 7) + (lane << 2) + (j & 3);
}

// ---------------------------------------------------------------------------
// Pack W [Ntot x Ktot] (row-major f32) into bf16 WMMA B-fragments for
// B[k][n] = W[n][k].  B layout (32x16): N = lane%16,
//   K = (j>>2)*16 + (lane/16)*8 + 2(j&3) + p;  dword = tile*256 + (j>>2)*128 + lane*4 + (j&3).
// ---------------------------------------------------------------------------
__global__ void pack_b_bf16(const float* __restrict__ W,
                            unsigned int* __restrict__ out,
                            int Ntot, int Ktot) {
    long long idx   = (long long)blockIdx.x * blockDim.x + threadIdx.x;
    long long total = (long long)Ntot * Ktot / 2;  // output dwords
    if (idx >= total) return;
    int nKt        = Ktot >> 5;
    int q          = (int)(idx & 3);
    int lane       = (int)((idx >> 2) & 31);
    int half       = (int)((idx >> 7) & 1);
    long long tile = idx >> 8;
    int ktile      = (int)(tile % nKt);
    int ntile      = (int)(tile / nKt);
    int j          = half * 4 + q;
    int n          = ntile * 16 + (lane & 15);
    int k          = ktile * 32 + ((j >> 2) << 4) + ((lane >> 4) << 3) + ((j & 3) << 1);
    const float* row = W + (long long)n * Ktot + k;
    out[idx] = pk_bf16(row[0], row[1]);
}

// Load a pre-packed fragment (A or B): two coalesced b128 loads per lane.
__device__ __forceinline__ void load_frag(const uint4* __restrict__ tileBase,
                                          int lane, Frag16& f) {
    uint4 lo = tileBase[lane];
    uint4 hi = tileBase[32 + lane];
    f.u[0] = lo.x; f.u[1] = lo.y; f.u[2] = lo.z; f.u[3] = lo.w;
    f.u[4] = hi.x; f.u[5] = hi.y; f.u[6] = hi.z; f.u[7] = hi.w;
}

// ---------------------------------------------------------------------------
// Gather token embedding rows and emit x as packed bf16 A-fragments.
// ---------------------------------------------------------------------------
__global__ void __launch_bounds__(256)
embed_pack_a(const float* __restrict__ embed,
             const int* __restrict__ seqs, int t, int start_flag,
             unsigned int* __restrict__ x_pk) {
    int idx = blockIdx.x * blockDim.x + threadIdx.x;  // B*E/2 = 16384
    if (idx >= BB * EE / 2) return;
    int m = idx / (EE / 2);
    int n = (idx % (EE / 2)) * 2;
    int tok = start_flag ? 1 : seqs[m * TT + t];
    const float* row = embed + (long long)tok * EE + n;
    x_pk[a_frag_dword(m, n, EE >> 5)] = pk_bf16(row[0], row[1]);
}

// ---------------------------------------------------------------------------
// gates = x @ Wih^T + h @ Whh^T + bias  (x, h pre-packed A-fragments)
// One wave per 16(M) x 32(N) super-tile: A fragment reused across 2 B tiles,
// two independent WMMA accumulator chains.  Grid: 128 blocks x 128 thr.
// ---------------------------------------------------------------------------
__global__ void __launch_bounds__(128)
lstm_gates_wmma(const unsigned int* __restrict__ x_pk,
                const unsigned int* __restrict__ h_pk,
                const unsigned int* __restrict__ Wih_pk,
                const unsigned int* __restrict__ Whh_pk,
                const float* __restrict__ bias,
                float* __restrict__ gates) {
    const int lane   = threadIdx.x & 31;
    const int wid    = blockIdx.x * 4 + (threadIdx.x >> 5);  // 0..511
    const int tileM  = wid & 3;                              // batch tiles (64/16)
    const int tileNN = wid >> 2;                             // 0..127 (2 N-tiles each)
    const int mbase  = tileM << 4;
    const int nbase  = tileNN << 5;

    v8f acc0, acc1;
    const float bv0 = bias[nbase + (lane & 15)];
    const float bv1 = bias[nbase + 16 + (lane & 15)];
#pragma unroll
    for (int r = 0; r < 8; ++r) { acc0[r] = bv0; acc1[r] = bv1; }

    // ---- x @ Wih^T  (K = E = 512, 16 k-tiles) ----
    {
        const int nKt = EE >> 5;
        const uint4* abase  = (const uint4*)x_pk  + (long long)tileM * nKt * 64;
        const uint4* bbase0 = (const uint4*)Wih_pk + (long long)(2 * tileNN)     * nKt * 64;
        const uint4* bbase1 = (const uint4*)Wih_pk + (long long)(2 * tileNN + 1) * nKt * 64;
#pragma unroll 4
        for (int kt = 0; kt < nKt; ++kt) {
            Frag16 a, b0, b1;
            load_frag(abase  + kt * 64, lane, a);
            load_frag(bbase0 + kt * 64, lane, b0);
            load_frag(bbase1 + kt * 64, lane, b1);
            if (kt + 1 < nKt) {
                __builtin_prefetch(bbase0 + (kt + 1) * 64 + lane, 0, 1);
                __builtin_prefetch(bbase1 + (kt + 1) * 64 + lane, 0, 1);
            }
            acc0 = __builtin_amdgcn_wmma_f32_16x16x32_bf16(
                false, a.v, false, b0.v, (short)0, acc0, false, false);
            acc1 = __builtin_amdgcn_wmma_f32_16x16x32_bf16(
                false, a.v, false, b1.v, (short)0, acc1, false, false);
        }
    }
    // ---- h @ Whh^T  (K = H = 1024, 32 k-tiles) ----
    {
        const int nKt = HH >> 5;
        const uint4* abase  = (const uint4*)h_pk  + (long long)tileM * nKt * 64;
        const uint4* bbase0 = (const uint4*)Whh_pk + (long long)(2 * tileNN)     * nKt * 64;
        const uint4* bbase1 = (const uint4*)Whh_pk + (long long)(2 * tileNN + 1) * nKt * 64;
#pragma unroll 4
        for (int kt = 0; kt < nKt; ++kt) {
            Frag16 a, b0, b1;
            load_frag(abase  + kt * 64, lane, a);
            load_frag(bbase0 + kt * 64, lane, b0);
            load_frag(bbase1 + kt * 64, lane, b1);
            if (kt + 1 < nKt) {
                __builtin_prefetch(bbase0 + (kt + 1) * 64 + lane, 0, 1);
                __builtin_prefetch(bbase1 + (kt + 1) * 64 + lane, 0, 1);
            }
            acc0 = __builtin_amdgcn_wmma_f32_16x16x32_bf16(
                false, a.v, false, b0.v, (short)0, acc0, false, false);
            acc1 = __builtin_amdgcn_wmma_f32_16x16x32_bf16(
                false, a.v, false, b1.v, (short)0, acc1, false, false);
        }
    }

    // C/D layout: VGPR r -> M = r + 8*(lane/16), N = lane%16
#pragma unroll
    for (int r = 0; r < 8; ++r) {
        int M = r + ((lane >> 4) << 3);
        long long rowOff = (long long)(mbase + M) * G4H;
        gates[rowOff + nbase + (lane & 15)]      = acc0[r];
        gates[rowOff + nbase + 16 + (lane & 15)] = acc1[r];
    }
}

// ---------------------------------------------------------------------------
// Elementwise LSTM cell update (torch gate order i, f, g, o); writes h
// directly as packed bf16 A-fragments.
// ---------------------------------------------------------------------------
__device__ __forceinline__ float sigmoidf_(float x) {
    return 1.0f / (1.0f + __expf(-x));
}

__global__ void __launch_bounds__(256)
lstm_update_pack(const float* __restrict__ gates,
                 float* __restrict__ c,
                 unsigned int* __restrict__ h_pk) {
    int idx = blockIdx.x * blockDim.x + threadIdx.x;  // B*H/2 = 32768
    if (idx >= BB * HH / 2) return;
    int m = idx / (HH / 2);
    int n = (idx % (HH / 2)) * 2;
    const float* g = gates + (long long)m * G4H;
    float hv[2];
#pragma unroll
    for (int p = 0; p < 2; ++p) {
        int np = n + p;
        float gi = sigmoidf_(g[np]);
        float gf = sigmoidf_(g[HH + np]);
        float gg = tanhf(g[2 * HH + np]);
        float go = sigmoidf_(g[3 * HH + np]);
        long long ci = (long long)m * HH + np;
        float cn = gf * c[ci] + gi * gg;
        c[ci] = cn;
        hv[p] = go * tanhf(cn);
    }
    h_pk[a_frag_dword(m, n, HH >> 5)] = pk_bf16(hv[0], hv[1]);
}

// ---------------------------------------------------------------------------
// logits = h @ out_W^T + out_b.  B stream (32 KB/wave, linear 1 KB tiles) is
// double-buffered through LDS with global_load_async_to_lds_b128; WMMA
// consumes tile kt from LDS while tile kt+1 streams in (ASYNCcnt pipelining).
// grid: 500 blocks x 128 thr; one wave per 16x16 tile of [64 x 8000].
// ---------------------------------------------------------------------------
__global__ void __launch_bounds__(128)
logits_wmma(const unsigned int* __restrict__ h_pk,
            const unsigned int* __restrict__ Wpk,
            const float* __restrict__ ob,
            float* __restrict__ out, long long rowStride) {
    __shared__ char smem[4 * 2 * 1024];   // per-wave double buffer (2 x 1 KB)

    const int lane  = threadIdx.x & 31;
    const int wv    = threadIdx.x >> 5;   // 0..3 = tileM
    const int tileN = blockIdx.x;         // 0..499
    const int mbase = wv << 4;
    const int nbase = tileN << 4;
    const int nKt   = HH >> 5;            // 32

    // LDS byte offset of this lane's 16B slot in buffer 0 (generic addr
    // truncates to the LDS offset per the aperture mapping).
    const unsigned lds0 =
        (unsigned)(size_t)(void*)&smem[wv * 2048 + lane * 16];

    const uint4* abase = (const uint4*)h_pk + (long long)wv * nKt * 64;
    const char*  bbase = (const char*)Wpk + (long long)tileN * nKt * 1024;

    v8f acc;
    const float bv = ob[nbase + (lane & 15)];
#pragma unroll
    for (int r = 0; r < 8; ++r) acc[r] = bv;

    // Prologue: stage k-tile 0 into buffer 0 (offset applies to LDS and
    // global address alike, so offset:512 moves both halves).
    {
        unsigned long long g = (unsigned long long)bbase + (unsigned)(lane * 16);
        asm volatile(
            "global_load_async_to_lds_b128 %0, %1, off\n\t"
            "global_load_async_to_lds_b128 %0, %1, off offset:512"
            :: "v"(lds0), "v"(g) : "memory");
    }

#pragma unroll 2
    for (int kt = 0; kt < nKt; ++kt) {
        if (kt + 1 < nKt) {
            unsigned lds_n = lds0 + (unsigned)(((kt + 1) & 1) << 10);
            unsigned long long g = (unsigned long long)bbase +
                                   (unsigned)((kt + 1) * 1024 + lane * 16);
            asm volatile(
                "global_load_async_to_lds_b128 %0, %1, off\n\t"
                "global_load_async_to_lds_b128 %0, %1, off offset:512\n\t"
                "s_wait_asynccnt 0x2"
                :: "v"(lds_n), "v"(g) : "memory");
        } else {
            asm volatile("s_wait_asynccnt 0x0" ::: "memory");
        }

        const uint4* bl = (const uint4*)(smem + wv * 2048 + ((kt & 1) << 10));
        Frag16 a, b;
        load_frag(abase + kt * 64, lane, a);
        uint4 lo = bl[lane];
        uint4 hi = bl[32 + lane];
        b.u[0] = lo.x; b.u[1] = lo.y; b.u[2] = lo.z; b.u[3] = lo.w;
        b.u[4] = hi.x; b.u[5] = hi.y; b.u[6] = hi.z; b.u[7] = hi.w;

        acc = __builtin_amdgcn_wmma_f32_16x16x32_bf16(
            false, a.v, false, b.v, (short)0, acc, false, false);
    }

#pragma unroll
    for (int r = 0; r < 8; ++r) {
        int M = r + ((lane >> 4) << 3);
        out[(long long)(mbase + M) * rowStride + nbase + (lane & 15)] = acc[r];
    }
}

__global__ void zero_u32(unsigned int* __restrict__ p, int n) {
    int i = blockIdx.x * blockDim.x + threadIdx.x;
    if (i < n) p[i] = 0u;
}

// ---------------------------------------------------------------------------
// Host launcher
// ---------------------------------------------------------------------------
extern "C" void kernel_launch(void* const* d_in, const int* in_sizes, int n_in,
                              void* d_out, int out_size, void* d_ws, size_t ws_size,
                              hipStream_t stream) {
    (void)in_sizes; (void)n_in; (void)out_size; (void)ws_size;

    const int*   seqs      = (const int*)d_in[1];
    const float* enc_embed = (const float*)d_in[2];
    const float* enc_Wih   = (const float*)d_in[3];
    const float* enc_Whh   = (const float*)d_in[4];
    const float* enc_b     = (const float*)d_in[5];
    const float* dec_embed = (const float*)d_in[6];
    const float* dec_Wih   = (const float*)d_in[7];
    const float* dec_Whh   = (const float*)d_in[8];
    const float* dec_b     = (const float*)d_in[9];
    const float* out_W     = (const float*)d_in[10];
    const float* out_b     = (const float*)d_in[11];
    float*       out       = (float*)d_out;

    // --- workspace carve-up (aligned to 256B) ---
    char* ws = (char*)d_ws;
    size_t off = 0;
    auto take = [&](size_t bytes) -> void* {
        void* p = ws + off;
        off = (off + bytes + 255) & ~(size_t)255;
        return p;
    };
    unsigned int* encWih_pk = (unsigned int*)take((size_t)G4H * EE * 2);  // 4 MB
    unsigned int* encWhh_pk = (unsigned int*)take((size_t)G4H * HH * 2);  // 8 MB
    unsigned int* decWih_pk = (unsigned int*)take((size_t)G4H * EE * 2);  // 4 MB
    unsigned int* decWhh_pk = (unsigned int*)take((size_t)G4H * HH * 2);  // 8 MB
    unsigned int* outW_pk   = (unsigned int*)take((size_t)VV  * HH * 2);  // 16 MB
    float*        gatesBuf  = (float*)take((size_t)BB * G4H * 4);         // 1 MB
    float*        cBuf      = (float*)take((size_t)BB * HH * 4);          // 256 KB
    unsigned int* hPk       = (unsigned int*)take((size_t)BB * HH / 2 * 4); // 128 KB
    unsigned int* xPk       = (unsigned int*)take((size_t)BB * EE / 2 * 4); //  64 KB

    // --- one-time weight conversion + WMMA-fragment pre-swizzle (bf16) ---
    auto packLaunch = [&](const float* W, unsigned int* dst, int Ntot, int Ktot) {
        long long dwords = (long long)Ntot * Ktot / 2;
        int blocks = (int)((dwords + 255) / 256);
        pack_b_bf16<<<blocks, 256, 0, stream>>>(W, dst, Ntot, Ktot);
    };
    packLaunch(enc_Wih, encWih_pk, G4H, EE);
    packLaunch(enc_Whh, encWhh_pk, G4H, HH);
    packLaunch(dec_Wih, decWih_pk, G4H, EE);
    packLaunch(dec_Whh, decWhh_pk, G4H, HH);
    packLaunch(out_W,   outW_pk,   VV,  HH);

    // --- zero initial hidden (packed) and cell state ---
    zero_u32<<<(BB * HH / 2 + 255) / 256, 256, 0, stream>>>(hPk, BB * HH / 2);
    zero_u32<<<(BB * HH + 255) / 256, 256, 0, stream>>>((unsigned int*)cBuf, BB * HH);

    const int updBlocks = (BB * HH / 2 + 255) / 256;
    const int embBlocks = (BB * EE / 2 + 255) / 256;

    // --- encoder: T steps ---
    for (int t = 0; t < TT; ++t) {
        embed_pack_a<<<embBlocks, 256, 0, stream>>>(enc_embed, seqs, t, 0, xPk);
        lstm_gates_wmma<<<128, 128, 0, stream>>>(
            xPk, hPk, encWih_pk, encWhh_pk, enc_b, gatesBuf);
        lstm_update_pack<<<updBlocks, 256, 0, stream>>>(gatesBuf, cBuf, hPk);
    }

    // --- decoder with teacher forcing: T-1 steps; token j = (j==0 ? START : seqs[:, j]) ---
    const long long rowStride = (long long)(TT - 1) * VV;
    for (int j = 0; j < TT - 1; ++j) {
        embed_pack_a<<<embBlocks, 256, 0, stream>>>(dec_embed, seqs, j,
                                                    (j == 0) ? 1 : 0, xPk);
        lstm_gates_wmma<<<128, 128, 0, stream>>>(
            xPk, hPk, decWih_pk, decWhh_pk, dec_b, gatesBuf);
        lstm_update_pack<<<updBlocks, 256, 0, stream>>>(gatesBuf, cBuf, hPk);
        logits_wmma<<<VV / 16, 128, 0, stream>>>(
            hPk, outW_pk, out_b, out + (long long)j * VV, rowStride);
    }
}
